// BitMLPBlock_64570538328618
// MI455X (gfx1250) — 8-bit matrix operand order repaired and checked by reading, compiled, not run on hardware
//
#include <hip/hip_runtime.h>

typedef __attribute__((ext_vector_type(8))) int v8i;

#define DIM   512
#define HID   2048
#define TPB   32
#define MT    2
#define NT1   (HID / 16)
#define KB1   (DIM / 64)
#define NT2   (DIM / 16)
#define KB2   (HID / 64)
#define WELEM (HID * DIM)


__global__ void zero_sums(float* p) {
    if (threadIdx.x < 2) p[threadIdx.x] = 0.0f;
}

__global__ void abs_sum_kernel(const float* __restrict__ w, int n,
                               float* __restrict__ acc) {
    __shared__ float red[256];
    float s = 0.0f;
    for (long i = (long)blockIdx.x * 256 + threadIdx.x; i < n;
         i += (long)gridDim.x * 256)
        s += fabsf(w[i]);
    red[threadIdx.x] = s;
    __syncthreads();
    for (int off = 128; off > 0; off >>= 1) {
        if (threadIdx.x < off) red[threadIdx.x] += red[threadIdx.x + off];
        __syncthreads();
    }
    if (threadIdx.x == 0) atomicAdd(acc, red[0]);
}

__global__ void tern_pack_kernel(const float* __restrict__ w,
                                 signed char* __restrict__ wp,
                                 int kblks, int rowlen,
                                 const float* __restrict__ sum, float invcnt) {
    long idx   = (long)blockIdx.x * 256 + threadIdx.x;
    int  b     = (int)(idx & 3);
    int  v     = (int)((idx >> 2) & 7);
    int  lane  = (int)((idx >> 5) & 31);
    long rest  = idx >> 10;
    int  kb    = (int)(rest % kblks);
    int  nt    = (int)(rest / kblks);
    int  n     = nt * 16 + (lane & 15);
    int  k     = kb * 64 + (v >> 2) * 32 + ((v >> 1) & 1) * 16 + (v & 1) * 4 + (lane >> 4) * 8 + b;
    float mean = fmaxf(sum[0] * invcnt, 1e-5f);
    float s    = 1.0f / mean;
    float t    = fminf(fmaxf(rintf(w[(long)n * rowlen + k] * s), -1.0f), 1.0f);
    wp[idx]    = (signed char)(int)t;
}

__global__ __launch_bounds__(256) void bitmlp_main(
    const float* __restrict__ x,  const float* __restrict__ b1,
    const float* __restrict__ b2, const signed char* __restrict__ w1p,
    const signed char* __restrict__ w2p, const float* __restrict__ sums,
    float* __restrict__ out) {
    __shared__ signed char s_xq[TPB][DIM];
    __shared__ signed char s_hq[TPB][HID];
    __shared__ int s_xmax[TPB];
    __shared__ int s_hmax[TPB];

    const int  tid    = threadIdx.x;
    const int  lane   = tid & 31;
    const int  wave   = tid >> 5;
    const long token0 = (long)blockIdx.x * TPB;

    const float w1f = fmaxf(sums[0] * (1.0f / WELEM), 1e-5f);
    const float w2f = fmaxf(sums[1] * (1.0f / WELEM), 1e-5f);

    if (tid < TPB) { s_xmax[tid] = 0; s_hmax[tid] = 0; }
    __syncthreads();

    {
        int t = tid >> 3, j = tid & 7;
        const float* xr = x + (token0 + t) * DIM + j * 64;
        float mx = 0.0f;
        for (int i = 0; i < 64; ++i) mx = fmaxf(mx, fabsf(xr[i]));
        atomicMax(&s_xmax[t], __float_as_int(mx));
    }
    __syncthreads();

    for (int e = tid; e < TPB * DIM; e += 256) {
        int t = e >> 9, d = e & (DIM - 1);
        float sc = 127.0f / fmaxf(__int_as_float(s_xmax[t]), 1e-5f);
        float q  = rintf(x[(token0 + t) * DIM + d] * sc);
        q = fminf(fmaxf(q, -128.0f), 127.0f);
        s_xq[t][d] = (signed char)(int)q;
    }
    __syncthreads();

    const int m_hi = (lane >> 4) * 8;

    v8i a1[MT][KB1];
    {
        int m   = lane & 15;
        int khi = (lane >> 4) * 8;
#pragma unroll
        for (int mt = 0; mt < MT; ++mt) {
#pragma unroll
            for (int kb = 0; kb < KB1; ++kb) {
#pragma unroll
                for (int v = 0; v < 8; ++v) {
                    int k = kb * 64 + (v >> 2) * 32 + ((v >> 1) & 1) * 16 +
                            (v & 1) * 4 + khi;
                    a1[mt][kb][v] = *(const int*)&s_xq[mt * 16 + m][k];
                }
            }
        }
    }

    float g[MT][16][8];
    float rmax[MT][8];
#pragma unroll
    for (int mt = 0; mt < MT; ++mt)
#pragma unroll
        for (int r = 0; r < 8; ++r) rmax[mt][r] = 0.0f;

#pragma unroll
    for (int nt = 0; nt < 16; ++nt) {
        const int ntg = wave * 16 + nt;
        if (nt < 15)
            __builtin_prefetch(
                w1p + (((long)((ntg + 1) * KB1) * 32 + lane) << 5), 0, 0);
        v8i acc[MT];
#pragma unroll
        for (int mt = 0; mt < MT; ++mt) acc[mt] = (v8i){0, 0, 0, 0, 0, 0, 0, 0};
#pragma unroll
        for (int kb = 0; kb < KB1; ++kb) {
            const v8i bfr =
                *(const v8i*)(w1p + (((long)(ntg * KB1 + kb) * 32 + lane) << 5));
#pragma unroll
            for (int mt = 0; mt < MT; ++mt)
                acc[mt] = __builtin_amdgcn_wmma_i32_16x16x64_iu8(
                    true, a1[mt][kb], true, bfr, acc[mt], false, false);
        }
#pragma unroll
        for (int mt = 0; mt < MT; ++mt) {
#pragma unroll
            for (int r = 0; r < 8; ++r) {
                int   m  = mt * 16 + r + m_hi;
                int   n  = ntg * 16 + (lane & 15);
                float xf = fmaxf(__int_as_float(s_xmax[m]), 1e-5f) *
                           (1.0f / 127.0f);
                float h  = (float)acc[mt][r] * xf * w1f + b1[n];
                float ge = 0.5f * h *
                           (1.0f + erff(h * 0.70710678118654752440f));
                g[mt][nt][r] = ge;
                rmax[mt][r]  = fmaxf(rmax[mt][r], fabsf(ge));
            }
        }
    }
#pragma unroll
    for (int mt = 0; mt < MT; ++mt)
#pragma unroll
        for (int r = 0; r < 8; ++r)
            atomicMax(&s_hmax[mt * 16 + r + m_hi], __float_as_int(rmax[mt][r]));
    __syncthreads();

#pragma unroll
    for (int mt = 0; mt < MT; ++mt) {
#pragma unroll
        for (int r = 0; r < 8; ++r) {
            int   m  = mt * 16 + r + m_hi;
            float sc = 127.0f / fmaxf(__int_as_float(s_hmax[m]), 1e-5f);
#pragma unroll
            for (int nt = 0; nt < 16; ++nt) {
                int   n = (wave * 16 + nt) * 16 + (lane & 15);
                float q =
                    fminf(fmaxf(rintf(g[mt][nt][r] * sc), -128.0f), 127.0f);
                s_hq[m][n] = (signed char)(int)q;
            }
        }
    }
    __syncthreads();

    v8i acc2[MT][4];
#pragma unroll
    for (int mt = 0; mt < MT; ++mt)
#pragma unroll
        for (int j = 0; j < 4; ++j) acc2[mt][j] = (v8i){0, 0, 0, 0, 0, 0, 0, 0};
    {
        int m   = lane & 15;
        int khi = (lane >> 4) * 8;
        for (int kb = 0; kb < KB2; ++kb) {
            v8i a[MT];
#pragma unroll
            for (int mt = 0; mt < MT; ++mt) {
#pragma unroll
                for (int v = 0; v < 8; ++v) {
                    int k = kb * 64 + (v >> 2) * 32 + ((v >> 1) & 1) * 16 +
                            (v & 1) * 4 + khi;
                    a[mt][v] = *(const int*)&s_hq[mt * 16 + m][k];
                }
            }
#pragma unroll
            for (int j = 0; j < 4; ++j) {
                const int ntg = wave * 4 + j;
                const v8i bfr = *(const v8i*)(w2p +
                    (((long)(ntg * KB2 + kb) * 32 + lane) << 5));
#pragma unroll
                for (int mt = 0; mt < MT; ++mt)
                    acc2[mt][j] = __builtin_amdgcn_wmma_i32_16x16x64_iu8(
                        true, a[mt], true, bfr, acc2[mt][j], false, false);
            }
        }
    }

#pragma unroll
    for (int mt = 0; mt < MT; ++mt) {
#pragma unroll
        for (int j = 0; j < 4; ++j) {
            const int ntg = wave * 4 + j;
#pragma unroll
            for (int r = 0; r < 8; ++r) {
                int   m  = mt * 16 + r + m_hi;
                int   n  = ntg * 16 + (lane & 15);
                float hf = fmaxf(__int_as_float(s_hmax[m]), 1e-5f) *
                           (1.0f / 127.0f);
                float v  = (float)acc2[mt][j][r] * hf * w2f + b2[n] +
                           x[(token0 + m) * DIM + n];
                out[(token0 + m) * DIM + n] = v;
            }
        }
    }
}

extern "C" void kernel_launch(void* const* d_in, const int* in_sizes, int n_in,
                              void* d_out, int out_size, void* d_ws,
                              size_t ws_size, hipStream_t stream) {
    const float* x  = (const float*)d_in[0];
    const float* w1 = (const float*)d_in[1];
    const float* b1 = (const float*)d_in[2];
    const float* w2 = (const float*)d_in[3];
    const float* b2 = (const float*)d_in[4];
    float*       out = (float*)d_out;

    float*       sums = (float*)d_ws;
    signed char* w1p  = (signed char*)d_ws + 256;
    signed char* w2p  = w1p + (size_t)NT1 * KB1 * 1024;

    zero_sums<<<1, 32, 0, stream>>>(sums);
    abs_sum_kernel<<<256, 256, 0, stream>>>(w1, WELEM, sums + 0);
    abs_sum_kernel<<<256, 256, 0, stream>>>(w2, WELEM, sums + 1);

    const long tot = (long)NT1 * KB1 * 1024;
    tern_pack_kernel<<<(int)((tot + 255) / 256), 256, 0, stream>>>(
        w1, w1p, KB1, DIM, sums + 0, 1.0f / WELEM);
    tern_pack_kernel<<<(int)((tot + 255) / 256), 256, 0, stream>>>(
        w2, w2p, KB2, HID, sums + 1, 1.0f / WELEM);

    const int tokens = in_sizes[0] / DIM;
    bitmlp_main<<<tokens / TPB, 256, 0, stream>>>(x, b1, b2, w1p, w2p, sums,
                                                  out);
}
